// DGN8_70428873720399
// MI455X (gfx1250) — matrix operand order repaired and checked by reading, not compiled, not run on hardware
//
#include <hip/hip_runtime.h>
#include <hip/hip_bf16.h>
#include <math.h>

typedef __attribute__((ext_vector_type(16))) _Float16 v16h;
typedef __attribute__((ext_vector_type(8)))  _Float16 v8h;
typedef __attribute__((ext_vector_type(8)))  float    v8f;

#define BB 4
#define TT 2048
#define DD 512
#define NW 8          // waves per workgroup (256 threads, wave32)

// ---------------------------------------------------------------------------
// Kernel A: row L2-normalize h -> xn (f16, row major)
// ---------------------------------------------------------------------------
__global__ __launch_bounds__(256) void dgn_norm(const float* __restrict__ h,
                                                _Float16* __restrict__ xn) {
  const int row = blockIdx.x;                // b*T + t
  const float* hp = h + (size_t)row * DD;
  __shared__ float red[NW];
  float ss = 0.f;
  for (int d = threadIdx.x; d < DD; d += 256) { float v = hp[d]; ss += v * v; }
  for (int off = 16; off > 0; off >>= 1) ss += __shfl_down(ss, off, 32);
  if ((threadIdx.x & 31) == 0) red[threadIdx.x >> 5] = ss;
  __syncthreads();
  if (threadIdx.x == 0) {
    float tot = 0.f;
    for (int i = 0; i < NW; ++i) tot += red[i];
    red[0] = 1.0f / fmaxf(sqrtf(tot), 1e-12f);
  }
  __syncthreads();
  const float rn = red[0];
  for (int d = threadIdx.x; d < DD; d += 256)
    xn[(size_t)row * DD + d] = (_Float16)(hp[d] * rn);
}

// ---------------------------------------------------------------------------
// Kernel B: fused causal Gram (WMMA f16 -> f32) + per-row top-K / bottom-K.
// One workgroup = 16 consecutive rows of one batch. Waves split column tiles.
// ---------------------------------------------------------------------------
__global__ __launch_bounds__(256) void dgn_select(const _Float16* __restrict__ xn,
                                                  int* __restrict__ topI,
                                                  int* __restrict__ topC,
                                                  int* __restrict__ botI,
                                                  int* __restrict__ botC,
                                                  int KS, int KC) {
  const int ttile = blockIdx.x;              // 0..T/16-1
  const int b     = blockIdx.y;              // batch
  const int t0    = ttile * 16;
  const int tid   = threadIdx.x;
  const int w     = tid >> 5;                // wave id 0..7
  const int lane  = tid & 31;
  const int lm    = lane & 15;
  const int half  = lane >> 4;

  __shared__ __align__(32) _Float16 As[16][DD];   // 16 KB: the 16 strip rows
  __shared__ float tileC[NW][16][16];             // 8 KB: per-wave sim tile
  __shared__ float mTV[16][NW][16];               // merge: top vals
  __shared__ int   mTI[16][NW][16];               // merge: top idx
  __shared__ float mBV[16][NW][8];                // merge: bot vals
  __shared__ int   mBI[16][NW][8];                // merge: bot idx

  const _Float16* xb = xn + (size_t)b * TT * DD;

  // stage A rows into LDS (8 halves per thread-iteration)
  for (int i = tid; i < 16 * DD / 8; i += 256) {
    const int row = (i * 8) / DD;
    const int col = (i * 8) % DD;
    *(v8h*)&As[row][col] = *(const v8h*)&xb[(size_t)(t0 + row) * DD + col];
  }
  __syncthreads();

  // per-lane running selection lists (lanes 0..15 each own one strip row)
  float topV[16]; int topIx[16];
  float botV[8];  int botIx[8];
  for (int i = 0; i < 16; ++i) { topV[i] = -3.0e9f; topIx[i] = -1; }
  for (int i = 0; i < 8;  ++i) { botV[i] =  3.0e9f; botIx[i] = -1; }

  const int maxJT = ttile;   // strictly-past: columns only up to the diagonal tile
  for (int jt = w; jt <= maxJT; jt += NW) {
    const int j0 = jt * 16;
    v8f c = {};
    // B fragment: lane = column j0+lm, one contiguous 32B chunk per K-step
    const _Float16* bp = xb + (size_t)(j0 + lm) * DD + half * 16;
#pragma unroll
    for (int ks = 0; ks < DD / 32; ++ks) {
      const int k0 = ks * 32;
      // A fragment: row lm, contiguous 16 halves k0+16*half.. (matches B)
      v8h a0 = *(const v8h*)&As[lm][k0 + 16 * half];
      v8h a1 = *(const v8h*)&As[lm][k0 + 16 * half + 8];
      v16h a;
#pragma unroll
      for (int e = 0; e < 8; ++e) { a[e] = a0[e]; a[8 + e] = a1[e]; }
      v16h bf = *(const v16h*)(bp + k0);
      c = __builtin_amdgcn_wmma_f32_16x16x32_f16(false, a, false, bf,
                                                 (short)0, c, false, false);
    }
    // spill tile (lane holds column lm of rows v + 8*half) to wave-private LDS
#pragma unroll
    for (int v = 0; v < 8; ++v) tileC[w][v + 8 * half][lm] = c[v];
    __asm__ volatile("s_wait_dscnt 0x0" ::: "memory");   // wave-internal LDS RAW

    if (lane < 16) {                         // lane r scans its row
      const int t = t0 + lane;
#pragma unroll
      for (int cc = 0; cc < 16; ++cc) {
        const int s = j0 + cc;
        if (s < t) {                         // strictly past only
          const float val = tileC[w][lane][cc];
          if (val > topV[KS - 1]) {          // descending insert
            int p = KS - 1;
            while (p > 0 && topV[p - 1] < val) {
              topV[p] = topV[p - 1]; topIx[p] = topIx[p - 1]; --p;
            }
            topV[p] = val; topIx[p] = s;
          }
          if (val < botV[KC - 1]) {          // ascending insert
            int p = KC - 1;
            while (p > 0 && botV[p - 1] > val) {
              botV[p] = botV[p - 1]; botIx[p] = botIx[p - 1]; --p;
            }
            botV[p] = val; botIx[p] = s;
          }
        }
      }
    }
  }

  // publish per-wave partial lists
  if (lane < 16) {
    for (int i = 0; i < KS; ++i) { mTV[lane][w][i] = topV[i]; mTI[lane][w][i] = topIx[i]; }
    for (int i = 0; i < KC; ++i) { mBV[lane][w][i] = botV[i]; mBI[lane][w][i] = botIx[i]; }
  }
  __syncthreads();

  // merge: threads 0..15 each finalize one row
  if (tid < 16) {
    const int r = tid;
    const int t = t0 + r;
    float fV[16]; int fI[16];
    for (int i = 0; i < 16; ++i) { fV[i] = -3.0e9f; fI[i] = -1; }
    for (int wv = 0; wv < NW; ++wv)
      for (int i = 0; i < KS; ++i) {
        const float val = mTV[r][wv][i]; const int ix = mTI[r][wv][i];
        if (ix >= 0 && val > fV[KS - 1]) {
          int p = KS - 1;
          while (p > 0 && fV[p - 1] < val) { fV[p] = fV[p - 1]; fI[p] = fI[p - 1]; --p; }
          fV[p] = val; fI[p] = ix;
        }
      }
    int cnt = 0;
    for (int i = 0; i < KS; ++i) if (fI[i] >= 0) ++cnt;
    topC[b * TT + t] = cnt;
    for (int i = 0; i < 16; ++i)
      topI[(size_t)(b * TT + t) * 16 + i] = (i < KS) ? fI[i] : -1;

    float gV[8]; int gI[8];
    for (int i = 0; i < 8; ++i) { gV[i] = 3.0e9f; gI[i] = -1; }
    for (int wv = 0; wv < NW; ++wv)
      for (int i = 0; i < KC; ++i) {
        const float val = mBV[r][wv][i]; const int ix = mBI[r][wv][i];
        if (ix < 0) continue;
        bool inTop = false;                  // ref excludes top-k positions
        for (int q = 0; q < KS; ++q) if (fI[q] == ix) inTop = true;
        if (inTop) continue;
        if (val < gV[KC - 1]) {
          int p = KC - 1;
          while (p > 0 && gV[p - 1] > val) { gV[p] = gV[p - 1]; gI[p] = gI[p - 1]; --p; }
          gV[p] = val; gI[p] = ix;
        }
      }
    int bcnt = 0;
    for (int i = 0; i < KC; ++i) if (gI[i] >= 0) ++bcnt;
    botC[b * TT + t] = bcnt;
    for (int i = 0; i < 8; ++i)
      botI[(size_t)(b * TT + t) * 8 + i] = (i < KC) ? gI[i] : -1;
  }
}

// ---------------------------------------------------------------------------
// Kernel C: gather-mean aggregate, blend, exact GELU, momentum update
// ---------------------------------------------------------------------------
__global__ __launch_bounds__(256) void dgn_update(const float* __restrict__ hin,
                                                  float* __restrict__ hout,
                                                  const int* __restrict__ topI,
                                                  const int* __restrict__ topC,
                                                  const int* __restrict__ botI,
                                                  const int* __restrict__ botC,
                                                  const float* __restrict__ gain,
                                                  const float* __restrict__ bias,
                                                  const float* __restrict__ log_mix,
                                                  const float* __restrict__ log_alpha,
                                                  const float* __restrict__ log_momentum,
                                                  int KC) {
  const int row = blockIdx.x;                // b*T + t
  const int t   = row & (TT - 1);
  const float mix   = 1.f / (1.f + expf(-log_mix[0]));
  const float alpha = 1.f / (1.f + expf(-log_alpha[0]));
  const float mom   = 1.f / (1.f + expf(-log_momentum[0]));
  const int tc = topC[row];
  int bc = KC - (TT - t);                    // #real bottom picks (ref semantics)
  if (bc < 0) bc = 0;
  const int bstored = botC[row];
  if (bc > bstored) bc = bstored;
  const float rpos = 1.f / fmaxf((float)tc, 1.f);
  const float rneg = 1.f / fmaxf((float)bc, 1.f);
  const float* hb = hin + ((size_t)(row >> 11)) * TT * DD;   // batch base

  for (int d = threadIdx.x; d < DD; d += 256) {
    float mp = 0.f, mn = 0.f;
    for (int i = 0; i < tc; ++i) {
      const int ix = topI[(size_t)row * 16 + i];
      mp += hb[(size_t)ix * DD + d];
    }
    for (int i = 0; i < bc; ++i) {
      const int ix = botI[(size_t)row * 8 + i];
      mn += hb[(size_t)ix * DD + d];
    }
    mp *= rpos; mn *= rneg;
    const float h0      = hin[(size_t)row * DD + d];
    const float ctx     = alpha * mp + (1.f - alpha) * mn;
    const float blended = mix * h0 + (1.f - mix) * ctx;
    const float pre     = blended * gain[d] + bias[d];
    const float g       = 0.5f * pre * (1.f + erff(pre * 0.70710678118654752f));
    hout[(size_t)row * DD + d] = mom * h0 + (1.f - mom) * g;
  }
}

// ---------------------------------------------------------------------------
// Kernel D: out = (h - x) * (softplus(log_scale) + 0.01)
// ---------------------------------------------------------------------------
__global__ __launch_bounds__(256) void dgn_final(const float* __restrict__ h,
                                                 const float* __restrict__ x,
                                                 const float* __restrict__ log_scale,
                                                 float* __restrict__ out, int n) {
  const int i = blockIdx.x * 256 + threadIdx.x;
  if (i < n) {
    const float v  = log_scale[0];
    const float sp = (v > 20.f) ? v : log1pf(expf(v));
    out[i] = (h[i] - x[i]) * (sp + 0.01f);
  }
}

// ---------------------------------------------------------------------------
extern "C" void kernel_launch(void* const* d_in, const int* in_sizes, int n_in,
                              void* d_out, int out_size, void* d_ws, size_t ws_size,
                              hipStream_t stream) {
  (void)in_sizes; (void)n_in; (void)out_size; (void)ws_size;
  const float* x            = (const float*)d_in[0];
  const float* gain         = (const float*)d_in[1];
  const float* bias         = (const float*)d_in[2];
  const float* log_mix      = (const float*)d_in[3];
  const float* log_alpha    = (const float*)d_in[4];
  const float* log_momentum = (const float*)d_in[5];
  const float* log_scale    = (const float*)d_in[6];
  float* out = (float*)d_out;

  const size_t N = (size_t)BB * TT * DD;
  char* ws = (char*)d_ws;
  float*    h_ws = (float*)ws;                                // 16 MB
  _Float16* xn   = (_Float16*)(ws + N * sizeof(float));       //  8 MB
  int* topI = (int*)(ws + N * sizeof(float) + N * sizeof(_Float16));
  int* botI = topI + (size_t)BB * TT * 16;
  int* topC = botI + (size_t)BB * TT * 8;
  int* botC = topC + (size_t)BB * TT;

  const int KSs[3] = {4, 8, 16};
  const int KCs[3] = {2, 4, 8};

  for (int r = 0; r < 3; ++r) {
    // h ping-pong: r0: x -> h_ws ; r1: h_ws -> out ; r2: out -> h_ws
    const float* hin  = (r == 0) ? x : ((r == 1) ? h_ws : out);
    float*       hout = (r == 1) ? out : h_ws;

    dgn_norm<<<BB * TT, 256, 0, stream>>>(hin, xn);
    dgn_select<<<dim3(TT / 16, BB), 256, 0, stream>>>(xn, topI, topC, botI, botC,
                                                      KSs[r], KCs[r]);
    dgn_update<<<BB * TT, 256, 0, stream>>>(hin, hout, topI, topC, botI, botC,
                                            gain + (size_t)r * DD,
                                            bias + (size_t)r * DD,
                                            log_mix + r, log_alpha + r,
                                            log_momentum, KCs[r]);
  }
  dgn_final<<<(int)((N + 255) / 256), 256, 0, stream>>>(h_ws, x, log_scale, out, (int)N);
}
